// MixingNetwork_12713103196368
// MI455X (gfx1250) — compile-verified, matrix operand order checked by reading, not run on hardware
//
#include <hip/hip_runtime.h>
#include <hip/hip_bf16.h>
#include <stdint.h>

// ---------------------------------------------------------------------------
// QMIX mixing network for MI455X (gfx1250, wave32, WMMA).
//
// Dominant math reformulated as one bf16 WMMA GEMM:
//   out[b,h] = sum_{k=s*64+n} (state[b,s]*q[b,n]) * W'[k,h],  K = 32768
// A (X) generated on the fly: loop-invariant q fragments (bf16) scaled by the
// state scalar with native v_pk_mul_bf16 (forced via inline asm -- clang
// otherwise promotes bf16 math to f32). W' repacked to bf16 once (16.8 MB,
// L2-resident). BM=32 per WG halves L2 B-traffic; H split over grid.y=4 with
// a deterministic partial-sum reduction kernel keeps 512 WGs in flight.
// ---------------------------------------------------------------------------

typedef __attribute__((ext_vector_type(16))) __bf16 bf16x16;
typedef __attribute__((ext_vector_type(8)))  __bf16 bf16x8;
typedef __attribute__((ext_vector_type(2)))  __bf16 bf16x2;
typedef __attribute__((ext_vector_type(8)))  float  f32x8;

#define B_TOT    4096
#define N_AG     64
#define HID      256
#define SDIM     512
#define KBIG     32768          // SDIM * N_AG
#define AUXN     576            // 256 out_w + 256 hid_b + 1 out_b + pad (16-mult)
#define AUXK     576            // 512 state-K + 64 q-K (folds b_hw)

// ws layout (bytes)
#define WS_WBF   0u                       // bf16 [256][32768]  = 16,777,216 B
#define WS_WAUX  16777216u                // bf16 [576][576]    =    663,552 B
#define WS_AUX   17440768u                // f32  [4096][576]   =  9,437,184 B
#define WS_PART  26877952u                // f32  [4096][4]     =     65,536 B

__device__ __forceinline__ f32x8 wmma_bf16(bf16x16 a, bf16x16 b, f32x8 c) {
  return __builtin_amdgcn_wmma_f32_16x16x32_bf16(false, a, false, b,
                                                 (short)0, c, false, false);
}

// Broadcast one float into a packed bf16 pair (lowers to v_cvt_pk_bf16_f32 v,f,f)
__device__ __forceinline__ uint32_t bf16_splat2(float f) {
  union { bf16x2 v; uint32_t u; } un;
  un.v[0] = (__bf16)f;
  un.v[1] = (__bf16)f;
  return un.u;
}

// A-fragment scale: 8x v_pk_mul_bf16 (native packed bf16; inline asm because
// clang promotes __bf16 vector arithmetic to f32 on this target).
__device__ __forceinline__ bf16x16 pk_scale(bf16x16 q, uint32_t svpk) {
  union { bf16x16 v; uint32_t u[8]; } in, out;
  in.v = q;
#pragma unroll
  for (int i = 0; i < 8; ++i)
    asm("v_pk_mul_bf16 %0, %1, %2" : "=v"(out.u[i]) : "v"(in.u[i]), "v"(svpk));
  return out.v;
}

// ---------------------------------------------------------------------------
// Kernel 1a: W_hw [512, 256*64] f32 -> W'[h][k=s*64+n] bf16 (row-major, K contig)
// ---------------------------------------------------------------------------
__global__ __launch_bounds__(256) void qmix_repack_whw(
    const float* __restrict__ W_hw, __bf16* __restrict__ Wbf) {
  const int tid = blockIdx.x * 256 + threadIdx.x;   // 0 .. 4,194,303 (pairs)
  const int np  = tid & 31;                         // n-pair within 64-wide row
  const int s   = (tid >> 5) & 511;
  const int h   = tid >> 14;                        // 0..255
  const float2 v = *(const float2*)(W_hw + (size_t)s * 16384 + h * 64 + np * 2);
  bf16x2 o; o[0] = (__bf16)v.x; o[1] = (__bf16)v.y;
  *(bf16x2*)(Wbf + (size_t)h * KBIG + s * 64 + np * 2) = o;
}

// ---------------------------------------------------------------------------
// Kernel 1b: aux weight pack: Waux[j][k] bf16, K-contiguous rows.
//   j<256   : out_w  column j  (W_ow[k,j])          k<512
//   256..511: hid_b  column    (W_hb[k,j-256]) and  k>=512 -> b_hw[(j-256)*64 + (k-512)]
//   j==512  : out_b  column    (W_ob[k])
//   else    : zero pad
// ---------------------------------------------------------------------------
__global__ __launch_bounds__(256) void qmix_repack_aux(
    const float* __restrict__ W_ow, const float* __restrict__ W_hb,
    const float* __restrict__ W_ob, const float* __restrict__ b_hw,
    __bf16* __restrict__ Waux) {
  const int tid = blockIdx.x * 256 + threadIdx.x;   // 0 .. 576*576-1
  const int k = tid % AUXK;
  const int j = tid / AUXK;
  float v = 0.0f;
  if (k < 512) {
    if (j < 256)      v = W_ow[(size_t)k * 256 + j];
    else if (j < 512) v = W_hb[(size_t)k * 256 + (j - 256)];
    else if (j == 512) v = W_ob[k];
  } else {
    if (j >= 256 && j < 512) v = b_hw[(size_t)(j - 256) * 64 + (k - 512)];
  }
  Waux[(size_t)j * AUXK + k] = (__bf16)v;
}

// ---------------------------------------------------------------------------
// Kernel 2: aux GEMM  M=4096, N=576, K=576 (bf16 WMMA, f32 accum)
//   A row b = [ state[b,0:512] | q[b,0:64] ]   (generated from f32 on the fly)
//   auxOut[b][j] = A @ Waux^T + bias
// ---------------------------------------------------------------------------
__global__ __launch_bounds__(128) void qmix_aux_gemm(
    const float* __restrict__ qv, const float* __restrict__ state,
    const __bf16* __restrict__ Waux,
    const float* __restrict__ b_ow, const float* __restrict__ b_hb,
    const float* __restrict__ b_ob, float* __restrict__ auxOut) {
  const int lane  = threadIdx.x & 31;
  const int wave  = threadIdx.x >> 5;            // 0..3
  const int m0    = blockIdx.x * 16;
  const int j0    = blockIdx.y * 64 + wave * 16; // 9*64 = 576 columns
  const int bA    = lane & 15;
  const int khalf = (lane >> 4) << 3;            // 0 or 8 (16-bit frag K split)

  const float*  srow = state + (size_t)(m0 + bA) * SDIM;
  const float*  qrow = qv    + (size_t)(m0 + bA) * N_AG;
  const __bf16* pB   = Waux  + (size_t)(j0 + (lane & 15)) * AUXK + khalf;

  f32x8 acc = {};
  for (int k0 = 0; k0 < AUXK; k0 += 32) {
    const float* arow = (k0 < 512) ? (srow + k0) : (qrow + (k0 - 512));
    const float4 a0 = *(const float4*)(arow + khalf);
    const float4 a1 = *(const float4*)(arow + khalf + 4);
    const float4 a2 = *(const float4*)(arow + khalf + 16);
    const float4 a3 = *(const float4*)(arow + khalf + 20);
    bf16x16 av;
    av[0]  = (__bf16)a0.x; av[1]  = (__bf16)a0.y; av[2]  = (__bf16)a0.z; av[3]  = (__bf16)a0.w;
    av[4]  = (__bf16)a1.x; av[5]  = (__bf16)a1.y; av[6]  = (__bf16)a1.z; av[7]  = (__bf16)a1.w;
    av[8]  = (__bf16)a2.x; av[9]  = (__bf16)a2.y; av[10] = (__bf16)a2.z; av[11] = (__bf16)a2.w;
    av[12] = (__bf16)a3.x; av[13] = (__bf16)a3.y; av[14] = (__bf16)a3.z; av[15] = (__bf16)a3.w;
    const bf16x8 c0 = *(const bf16x8*)(pB + k0);
    const bf16x8 c1 = *(const bf16x8*)(pB + k0 + 16);
    bf16x16 bv;
#pragma unroll
    for (int i = 0; i < 8; ++i) { bv[i] = c0[i]; bv[i + 8] = c1[i]; }
    acc = wmma_bf16(av, bv, acc);
  }

  const int j = j0 + (lane & 15);
  float bias;
  if (j < 256)       bias = b_ow[j];
  else if (j < 512)  bias = b_hb[j - 256];
  else if (j == 512) bias = b_ob[0];
  else               bias = 0.0f;
#pragma unroll
  for (int r = 0; r < 8; ++r) {
    const int b = m0 + r + ((lane < 16) ? 0 : 8);   // C/D: VGPR r -> M=r (+8 hi half)
    auxOut[(size_t)b * AUXN + j] = acc[r] + bias;
  }
}

// ---------------------------------------------------------------------------
// Kernel 3: main fused GEMM + mix (partials).
//   Grid (128, 4): WG = 4 waves = 32 batch rows x 64 h columns.
//   Wave w -> one 16-col h tile, two 16-row M tiles sharing each B fragment.
//   Per wave: 512 s x 2 nb x 2 m = 2048 v_wmma_f32_16x16x32_bf16.
//   A-gen per s: 2x v_cvt_pk_bf16_f32 (splat) + 4x8 v_pk_mul_bf16.
// ---------------------------------------------------------------------------
__global__ __launch_bounds__(128) void qmix_main(
    const float* __restrict__ qv, const float* __restrict__ state,
    const __bf16* __restrict__ Wbf, const float* __restrict__ auxOut,
    float* __restrict__ partials) {
  __shared__ float sums[4][32];
  const int lane  = threadIdx.x & 31;
  const int wave  = threadIdx.x >> 5;              // 0..3
  const int b0    = blockIdx.x * 32;
  const int bA    = lane & 15;                     // A-fragment M row
  const int khalf = (lane >> 4) << 3;              // 0 or 8

  // Loop-invariant q fragments for both M tiles, pre-converted to bf16.
  // K-slot pattern per lane: klocal = {khalf..khalf+7, khalf+16..khalf+23}
  bf16x16 qlo[2], qhi[2];
#pragma unroll
  for (int m = 0; m < 2; ++m) {
    const float* qrow = qv + (size_t)(b0 + m * 16 + bA) * N_AG;
#pragma unroll
    for (int i = 0; i < 8; ++i) {
      qlo[m][i]     = (__bf16)qrow[khalf + i];
      qlo[m][i + 8] = (__bf16)qrow[16 + khalf + i];
      qhi[m][i]     = (__bf16)qrow[32 + khalf + i];
      qhi[m][i + 8] = (__bf16)qrow[48 + khalf + i];
    }
  }
  const float* srow0 = state + (size_t)(b0 + bA) * SDIM;
  const float* srow1 = srow0 + (size_t)16 * SDIM;
  const int     hcol = blockIdx.y * 64 + wave * 16 + (lane & 15); // B/D column
  const __bf16* pB   = Wbf + (size_t)hcol * KBIG + khalf;

  f32x8 acc0 = {}, acc1 = {};
  for (int s0 = 0; s0 < SDIM; s0 += 4) {
    const float4 sa = *(const float4*)(srow0 + s0);
    const float4 sb = *(const float4*)(srow1 + s0);
#pragma unroll
    for (int si = 0; si < 4; ++si) {
      const float sf0 = (si == 0) ? sa.x : (si == 1) ? sa.y : (si == 2) ? sa.z : sa.w;
      const float sf1 = (si == 0) ? sb.x : (si == 1) ? sb.y : (si == 2) ? sb.z : sb.w;
      const uint32_t sp0 = bf16_splat2(sf0);       // v_cvt_pk_bf16_f32
      const uint32_t sp1 = bf16_splat2(sf1);
      const int kbase = (s0 + si) * 64;
#pragma unroll
      for (int nb = 0; nb < 2; ++nb) {
        const int ko = kbase + nb * 32;
        const bf16x8 c0 = *(const bf16x8*)(pB + ko);
        const bf16x8 c1 = *(const bf16x8*)(pB + ko + 16);
        bf16x16 bv;
#pragma unroll
        for (int i = 0; i < 8; ++i) { bv[i] = c0[i]; bv[i + 8] = c1[i]; }
        // native packed bf16 scale of loop-invariant q fragments -> A
        const bf16x16 a0 = pk_scale(nb ? qhi[0] : qlo[0], sp0);
        const bf16x16 a1 = pk_scale(nb ? qhi[1] : qlo[1], sp1);
        acc0 = wmma_bf16(a0, bv, acc0);
        acc1 = wmma_bf16(a1, bv, acc1);
      }
    }
  }

  // Epilogue: hid = ELU(acc + hid_b); partial = sum_h hid * out_w  (this WG's 64 h)
#pragma unroll
  for (int m = 0; m < 2; ++m) {
#pragma unroll
    for (int r = 0; r < 8; ++r) {
      const int b = b0 + m * 16 + r + ((lane < 16) ? 0 : 8);
      const float* arow = auxOut + (size_t)b * AUXN;
      const float acv = (m == 0) ? acc0[r] : acc1[r];
      float v = acv + arow[256 + hcol];
      float e = (v > 0.0f) ? v : expm1f(v);
      float tot = e * arow[hcol];
      // reduce over the 16 lanes sharing this b (different h)
      tot += __shfl_xor(tot, 1);
      tot += __shfl_xor(tot, 2);
      tot += __shfl_xor(tot, 4);
      tot += __shfl_xor(tot, 8);
      if ((lane & 15) == 0)
        sums[wave][m * 16 + r + ((lane < 16) ? 0 : 8)] = tot;
    }
  }
  __syncthreads();
  if (threadIdx.x < 32) {
    const float p = sums[0][threadIdx.x] + sums[1][threadIdx.x] +
                    sums[2][threadIdx.x] + sums[3][threadIdx.x];
    partials[(size_t)(b0 + threadIdx.x) * 4 + blockIdx.y] = p;
  }
}

// ---------------------------------------------------------------------------
// Kernel 4: final deterministic reduction over the 4 h-groups, + out_b.
// ---------------------------------------------------------------------------
__global__ __launch_bounds__(256) void qmix_final(
    const float* __restrict__ partials, const float* __restrict__ auxOut,
    float* __restrict__ out) {
  const int b = blockIdx.x * 256 + threadIdx.x;    // 0..4095
  const float4 p = *(const float4*)(partials + (size_t)b * 4);
  out[b] = p.x + p.y + p.z + p.w + auxOut[(size_t)b * AUXN + 512];
}

// ---------------------------------------------------------------------------
extern "C" void kernel_launch(void* const* d_in, const int* in_sizes, int n_in,
                              void* d_out, int out_size, void* d_ws, size_t ws_size,
                              hipStream_t stream) {
  const float* qv    = (const float*)d_in[0];
  const float* state = (const float*)d_in[1];
  const float* W_hw  = (const float*)d_in[2];
  const float* b_hw  = (const float*)d_in[3];
  const float* W_ow  = (const float*)d_in[4];
  const float* b_ow  = (const float*)d_in[5];
  const float* W_hb  = (const float*)d_in[6];
  const float* b_hb  = (const float*)d_in[7];
  const float* W_ob  = (const float*)d_in[8];
  const float* b_ob  = (const float*)d_in[9];
  float* out = (float*)d_out;

  char*   ws       = (char*)d_ws;
  __bf16* Wbf      = (__bf16*)(ws + WS_WBF);
  __bf16* Waux     = (__bf16*)(ws + WS_WAUX);
  float*  auxOut   = (float*)(ws + WS_AUX);
  float*  partials = (float*)(ws + WS_PART);

  // 1) weight repacks (f32 -> bf16, WMMA-friendly K-contiguous layouts)
  qmix_repack_whw<<<16384, 256, 0, stream>>>(W_hw, Wbf);          // 4.19M pairs
  qmix_repack_aux<<<1296, 256, 0, stream>>>(W_ow, W_hb, W_ob, b_hw, Waux);
  // 2) hypernet aux GEMM -> out_w / hid_b(+q.b_hw) / out_b  (with biases)
  qmix_aux_gemm<<<dim3(256, 9), 128, 0, stream>>>(qv, state, Waux,
                                                  b_ow, b_hb, b_ob, auxOut);
  // 3) fused main GEMM (68.7 GFLOP) + ELU mix -> per-h-group partials
  qmix_main<<<dim3(128, 4), 128, 0, stream>>>(qv, state, Wbf, auxOut, partials);
  // 4) deterministic final reduction + out_b
  qmix_final<<<16, 256, 0, stream>>>(partials, auxOut, out);
}
